// OmniMoEExperts_75514114998537
// MI455X (gfx1250) — compile-verified, matrix operand order checked by reading, not run on hardware
//
#include <hip/hip_runtime.h>
#include <hip/hip_bf16.h>

#define T_TOK  2048
#define HDIM   1024
#define IDIM   768
#define TWO_I  1536
#define NEXP   16
#define TOPK   4

#define BM     128            // tokens per block tile
#define LDACT  (IDIM + 8)     // activation LDS row stride (halves), 16B-aligned rows
#define LDST   40             // staging LDS row stride (halves) = 80B, 16B-aligned

typedef __attribute__((ext_vector_type(16))) __bf16 v16bf;
typedef __attribute__((ext_vector_type(8)))  __bf16 v8bf;
typedef __attribute__((ext_vector_type(8)))  float  v8f;
typedef __attribute__((ext_vector_type(4)))  int    v4i;

#define BF16MMA(a, b, c) \
  __builtin_amdgcn_wmma_f32_16x16x32_bf16(false, (a), false, (b), (short)0, (c), false, false)

// ---------------- CDNA5 async global->LDS (guarded) ----------------
#if defined(__HIP_DEVICE_COMPILE__) && __has_builtin(__builtin_amdgcn_global_load_async_to_lds_b128)
#define HAVE_ASYNC_LDS 1
#endif

__device__ __forceinline__ void cp16_g2l(void* lds, const void* g) {
#ifdef HAVE_ASYNC_LDS
  __builtin_amdgcn_global_load_async_to_lds_b128(
      (__attribute__((address_space(1))) v4i*)g,
      (__attribute__((address_space(3))) v4i*)lds, 0, 0);
#else
  *(uint4*)lds = *(const uint4*)g;
#endif
}

__device__ __forceinline__ void wait_async0() {
#ifdef HAVE_ASYNC_LDS
#if __has_builtin(__builtin_amdgcn_s_wait_asynccnt)
  __builtin_amdgcn_s_wait_asynccnt(0);
#else
  asm volatile("s_wait_asynccnt 0" ::: "memory");
#endif
#endif
}

// Load a 16x32 bf16 A/B fragment from K-contiguous row-major LDS tile.
// p points at element [row][k0 + base] where base = (lane<16)?0:8.
__device__ __forceinline__ v16bf ldfrag(const __bf16* p) {
  union { v16bf v; v8bf h[2]; } u;
  u.h[0] = *(const v8bf*)p;         // K = base .. base+7
  u.h[1] = *(const v8bf*)(p + 16);  // K = base+16 .. base+23
  return u.v;
}

// Convert 8 consecutive fp32 -> 8 bf16, store 16B to LDS.
__device__ __forceinline__ void stage8(__bf16* dst, const float* src) {
  float4 f0 = *(const float4*)src;
  float4 f1 = *(const float4*)(src + 4);
  union { __bf16 b[8]; uint4 u; } o;
  o.b[0] = (__bf16)f0.x; o.b[1] = (__bf16)f0.y;
  o.b[2] = (__bf16)f0.z; o.b[3] = (__bf16)f0.w;
  o.b[4] = (__bf16)f1.x; o.b[5] = (__bf16)f1.y;
  o.b[6] = (__bf16)f1.z; o.b[7] = (__bf16)f1.w;
  *(uint4*)dst = o.u;
}

// ---------------- kernel 0: fp32 -> bf16 activation convert ----------------
__global__ __launch_bounds__(256) void cvt_x_kernel(const float* __restrict__ x,
                                                    __bf16* __restrict__ xb) {
  int i = (blockIdx.x * 256 + threadIdx.x) * 4;
  float4 f = *(const float4*)(x + i);
  union { __bf16 b[4]; uint2 u; } o;
  o.b[0] = (__bf16)f.x; o.b[1] = (__bf16)f.y;
  o.b[2] = (__bf16)f.z; o.b[3] = (__bf16)f.w;
  *(uint2*)(xb + i) = o.u;
}

// ---------------- kernel 1: routing (compact per-expert token lists) --------
__global__ __launch_bounds__(256) void route_kernel(const int* __restrict__ tki,
                                                    const float* __restrict__ tkw,
                                                    int* __restrict__ counts,
                                                    int* __restrict__ tok_ids,
                                                    float* __restrict__ tok_wts) {
  int t = blockIdx.x * 256 + threadIdx.x;
  if (t >= T_TOK) return;
  int   e0[TOPK];
  float w0[TOPK];
#pragma unroll
  for (int k = 0; k < TOPK; ++k) { e0[k] = tki[t * TOPK + k]; w0[k] = tkw[t * TOPK + k]; }
#pragma unroll
  for (int k = 0; k < TOPK; ++k) {
    int e = e0[k];
    bool first = true;
#pragma unroll
    for (int j = 0; j < TOPK; ++j) if (j < k && e0[j] == e) first = false;
    if (!first) continue;
    float w = 0.0f;
#pragma unroll
    for (int j = 0; j < TOPK; ++j) if (j >= k && e0[j] == e) w += w0[j];
    int pos = atomicAdd(&counts[e], 1);
    tok_ids[e * T_TOK + pos] = t;
    tok_wts[e * T_TOK + pos] = w;
  }
}

// ---------------- kernel 2: fused expert (GEMM1 -> SiLU*up -> GEMM2) --------
__global__ __launch_bounds__(256) void moe_expert_kernel(
    const __bf16* __restrict__ Xb, const float* __restrict__ Wgu,
    const float* __restrict__ Wd, const int* __restrict__ counts,
    const int* __restrict__ tok_ids, const float* __restrict__ tok_wts,
    float* __restrict__ out) {
  __shared__ __bf16 sh_act[BM * LDACT];     // [128][768+8] bf16 activation tile
  __shared__ __bf16 sh_a2[2][BM * LDST];    // X staging, double buffered
  __shared__ __bf16 sh_b[4][64 * LDST];     // W staging: p1 {gate,up}x2buf; p2 uses [0],[2]
  __shared__ int    sh_tok[BM];
  __shared__ float  sh_w[BM];

  const int e   = blockIdx.y;
  const int cnt = counts[e];
  const int m0  = blockIdx.x * BM;
  if (m0 >= cnt) return;

  const int tid  = threadIdx.x;
  const int lane = tid & 31;
  const int wave = tid >> 5;
  const int mw   = wave & 3;   // wave's 32-row band
  const int nw   = wave >> 2;  // wave's 32-col band within 64-col chunk
  const int lr   = lane & 15;
  const int base = (lane < 16) ? 0 : 8;
  const int moff = (lane >= 16) ? 8 : 0;

  if (tid < BM) {
    int idx = m0 + tid;
    if (idx < cnt) { sh_tok[tid] = tok_ids[e * T_TOK + idx]; sh_w[tid] = tok_wts[e * T_TOK + idx]; }
    else           { sh_tok[tid] = 0;                        sh_w[tid] = 0.0f; }
  }
  __syncthreads();

  const float* wgu_e = Wgu + (size_t)e * TWO_I * HDIM;
  const float* wd_e  = Wd  + (size_t)e * HDIM * IDIM;

  const int arow = tid >> 1;        // 0..127, A staging row
  const int aseg = (tid & 1) * 16;  // halves
  const int brow = tid >> 2;        // 0..63, B staging row
  const int bseg = (tid & 3) * 8;   // floats/halves

  // ---- staging helpers (per-thread slices) ----
  auto stage_p1 = [&](int jn, int kk, int b) {
    const __bf16* g = Xb + (size_t)sh_tok[arow] * HDIM + kk + aseg;
    __bf16* l = &sh_a2[b][arow * LDST + aseg];
    cp16_g2l(l, g);
    cp16_g2l(l + 8, g + 8);
    const float* gg = wgu_e + (size_t)(jn + brow) * HDIM + kk + bseg;
    const float* gu = wgu_e + (size_t)(IDIM + jn + brow) * HDIM + kk + bseg;
    __builtin_prefetch(gg + 32, 0, 1);
    __builtin_prefetch(gu + 32, 0, 1);
    stage8(&sh_b[2 * b][brow * LDST + bseg], gg);
    stage8(&sh_b[2 * b + 1][brow * LDST + bseg], gu);
  };
  auto stage_p2 = [&](int nc, int ki, int b) {
    const float* gd = wd_e + (size_t)(nc + brow) * IDIM + ki + bseg;
    __builtin_prefetch(gd + 32, 0, 1);
    stage8(&sh_b[2 * b][brow * LDST + bseg], gd);
  };

  // ================= Phase 1: h = silu(X Wg^T) * (X Wu^T) =================
  int buf = 0;
  stage_p1(0, 0, 0);  // prologue
  for (int jn = 0; jn < IDIM; jn += 64) {
    v8f accg[2][2] = {};
    v8f accu[2][2] = {};
    for (int kk = 0; kk < HDIM; kk += 32) {
      wait_async0();
      __syncthreads();  // buffer `buf` ready; previous buffer's readers done

      // issue next tile's staging (other buffer) before computing
      int nk = kk + 32, njn = jn;
      if (nk == HDIM) { nk = 0; njn += 64; }
      if (njn < IDIM) stage_p1(njn, nk, buf ^ 1);

      const __bf16* sa = sh_a2[buf];
      const __bf16* sg = sh_b[2 * buf];
      const __bf16* su = sh_b[2 * buf + 1];
      v16bf a0  = ldfrag(&sa[(mw * 32 + lr) * LDST + base]);
      v16bf a1  = ldfrag(&sa[(mw * 32 + 16 + lr) * LDST + base]);
      v16bf bg0 = ldfrag(&sg[(nw * 32 + lr) * LDST + base]);
      v16bf bg1 = ldfrag(&sg[(nw * 32 + 16 + lr) * LDST + base]);
      v16bf bu0 = ldfrag(&su[(nw * 32 + lr) * LDST + base]);
      v16bf bu1 = ldfrag(&su[(nw * 32 + 16 + lr) * LDST + base]);

      accg[0][0] = BF16MMA(a0, bg0, accg[0][0]);
      accg[0][1] = BF16MMA(a0, bg1, accg[0][1]);
      accg[1][0] = BF16MMA(a1, bg0, accg[1][0]);
      accg[1][1] = BF16MMA(a1, bg1, accg[1][1]);
      accu[0][0] = BF16MMA(a0, bu0, accu[0][0]);
      accu[0][1] = BF16MMA(a0, bu1, accu[0][1]);
      accu[1][0] = BF16MMA(a1, bu0, accu[1][0]);
      accu[1][1] = BF16MMA(a1, bu1, accu[1][1]);
      buf ^= 1;
    }
    // epilogue: silu(gate)*up -> sh_act (bf16); fast rcp (result is bf16 anyway)
#pragma unroll
    for (int tm = 0; tm < 2; ++tm)
#pragma unroll
      for (int tn = 0; tn < 2; ++tn) {
        int mb = mw * 32 + tm * 16 + moff;
        int nb = jn + nw * 32 + tn * 16 + lr;
#pragma unroll
        for (int r = 0; r < 8; ++r) {
          float gv = accg[tm][tn][r];
          float sg_ = __builtin_amdgcn_rcpf(1.0f + __expf(-gv));
          float hv = gv * sg_ * accu[tm][tn][r];
          sh_act[(mb + r) * LDACT + nb] = (__bf16)hv;
        }
      }
  }
  __syncthreads();  // sh_act complete; phase-1 staging readers done

  // ================= Phase 2: out += w_tok * (h Wd^T) =================
  buf = 0;
  stage_p2(0, 0, 0);  // prologue (safe: after barrier, writes sh_b[0] only)
  for (int nc = 0; nc < HDIM; nc += 64) {
    v8f acc[2][2] = {};
    for (int ki = 0; ki < IDIM; ki += 32) {
      __syncthreads();  // buffer `buf` ready

      int nk = ki + 32, nnc = nc;
      if (nk == IDIM) { nk = 0; nnc += 64; }
      if (nnc < HDIM) stage_p2(nnc, nk, buf ^ 1);

      const __bf16* sd = sh_b[2 * buf];
      v16bf a0 = ldfrag(&sh_act[(mw * 32 + lr) * LDACT + ki + base]);
      v16bf a1 = ldfrag(&sh_act[(mw * 32 + 16 + lr) * LDACT + ki + base]);
      v16bf b0 = ldfrag(&sd[(nw * 32 + lr) * LDST + base]);
      v16bf b1 = ldfrag(&sd[(nw * 32 + 16 + lr) * LDST + base]);

      acc[0][0] = BF16MMA(a0, b0, acc[0][0]);
      acc[0][1] = BF16MMA(a0, b1, acc[0][1]);
      acc[1][0] = BF16MMA(a1, b0, acc[1][0]);
      acc[1][1] = BF16MMA(a1, b1, acc[1][1]);
      buf ^= 1;
    }
#pragma unroll
    for (int tm = 0; tm < 2; ++tm)
#pragma unroll
      for (int tn = 0; tn < 2; ++tn) {
        int mb = mw * 32 + tm * 16 + moff;
        int nb = nc + nw * 32 + tn * 16 + lr;
#pragma unroll
        for (int r = 0; r < 8; ++r) {
          int m = mb + r;
          float v = acc[tm][tn][r] * sh_w[m];
          atomicAdd(&out[(size_t)sh_tok[m] * HDIM + nb], v);
        }
      }
  }
}

// ---------------- host launcher ----------------
extern "C" void kernel_launch(void* const* d_in, const int* in_sizes, int n_in,
                              void* d_out, int out_size, void* d_ws, size_t ws_size,
                              hipStream_t stream) {
  const float* hidden = (const float*)d_in[0];
  const int*   tki    = (const int*)d_in[1];
  const float* tkw    = (const float*)d_in[2];
  const float* gup    = (const float*)d_in[3];
  const float* dwn    = (const float*)d_in[4];
  float* out = (float*)d_out;
  (void)in_sizes; (void)n_in; (void)out_size; (void)ws_size;

  char* ws = (char*)d_ws;
  int*    counts  = (int*)ws;                                        // 256 B slot
  int*    tok_ids = (int*)(ws + 256);                                // E*T ints
  float*  tok_wts = (float*)(ws + 256 + (size_t)NEXP * T_TOK * 4);   // E*T floats
  __bf16* Xb      = (__bf16*)(ws + 256 + 2 * (size_t)NEXP * T_TOK * 4); // T*H bf16

  (void)hipMemsetAsync(d_out, 0, (size_t)T_TOK * HDIM * sizeof(float), stream);
  (void)hipMemsetAsync(counts, 0, 256, stream);

  cvt_x_kernel<<<(T_TOK * HDIM / 4) / 256, 256, 0, stream>>>(hidden, Xb);
  route_kernel<<<(T_TOK + 255) / 256, 256, 0, stream>>>(tki, tkw, counts, tok_ids, tok_wts);

  dim3 grid(T_TOK / BM, NEXP);
  moe_expert_kernel<<<grid, 256, 0, stream>>>(Xb, gup, dwn, counts, tok_ids, tok_wts, out);
}
